// Transcoder_592705487229
// MI455X (gfx1250) — matrix operand order repaired and checked by reading, not compiled, not run on hardware
//
#include <hip/hip_runtime.h>

// ---------------------------------------------------------------------------
// Fused SAE encoder: pre = x @ W_enc^T + b ; relu ; per-token top-64
//   x: (2048 x 2048) f32, W_enc: (65536 x 2048) f32, b: (65536) f32
// Plan (MI455X / gfx1250, wave32, WMMA):
//   0) one-shot f32 -> bf16 (RNE) conversion of x and W into d_ws
//      (halves streamed bytes, enables v_wmma_f32_16x16x32_bf16)
//   1) GEMM+bias+relu+chunk-local top-64  -> candidates in d_ws
//   2) per-token merge of 32*64 candidates -> exact top-64 (sorted desc)
// ---------------------------------------------------------------------------

#define S        2048      // tokens
#define DM       2048      // d_model (K)
#define NF       65536     // features (N)
#define KSEL     64        // top-k
#define M_TILE   32        // tokens per workgroup
#define N_STEP   256       // features per inner slab (8 waves * 32)
#define N_CHUNK  2048      // features per workgroup
#define N_SUB    (N_CHUNK / N_STEP)   // 8
#define CHUNKS   (NF / N_CHUNK)       // 32
#define NEG_INF  (-3.402823466e38f)

typedef __attribute__((ext_vector_type(16))) __bf16 v16bf;
typedef __attribute__((ext_vector_type(8)))  __bf16 v8bf;
typedef __attribute__((ext_vector_type(8)))  float  v8f;

__device__ __forceinline__ __bf16 f32_to_bf16_rne(float f) {
  unsigned u = __builtin_bit_cast(unsigned, f);
  u += 0x7FFFu + ((u >> 16) & 1u);             // round-to-nearest-even
  unsigned short h = (unsigned short)(u >> 16);
  return __builtin_bit_cast(__bf16, h);
}

// -------------------- stage 0: f32 -> bf16 convert (streaming) -------------
__global__ __launch_bounds__(256) void cvt_f32_bf16(const float* __restrict__ in,
                                                    __bf16* __restrict__ out,
                                                    long n) {
  long base = (((long)blockIdx.x * 256) + threadIdx.x) * 8;
  if (base + 8 > n) return;
  const float4* p = reinterpret_cast<const float4*>(in + base);
  float4 a = p[0];
  float4 b = p[1];
  v8bf o;
  o[0] = f32_to_bf16_rne(a.x); o[1] = f32_to_bf16_rne(a.y);
  o[2] = f32_to_bf16_rne(a.z); o[3] = f32_to_bf16_rne(a.w);
  o[4] = f32_to_bf16_rne(b.x); o[5] = f32_to_bf16_rne(b.y);
  o[6] = f32_to_bf16_rne(b.z); o[7] = f32_to_bf16_rne(b.w);
  *reinterpret_cast<v8bf*>(out + base) = o;
}

// A fragment (16x32 bf16): lane l -> row m0+(l&15);
//   lanes 0-15 hold K 0..15, lanes 16-31 hold K 16..31 (contiguous per lane)
__device__ __forceinline__ v16bf load_a_frag(const __bf16* p) {
  v8bf lo = *reinterpret_cast<const v8bf*>(p);        // K off +0..7
  v8bf hi = *reinterpret_cast<const v8bf*>(p + 8);    // K off +8..15
  return __builtin_shufflevector(lo, hi, 0,1,2,3,4,5,6,7,8,9,10,11,12,13,14,15);
}

// B fragment (32x16 bf16): lane l -> col fbase+(l&15);
//   lanes 0-15 hold K 0..15, lanes 16-31 hold K 16..31 (contiguous per lane)
__device__ __forceinline__ v16bf load_b_frag(const __bf16* p) {
  v8bf lo = *reinterpret_cast<const v8bf*>(p);
  v8bf hi = *reinterpret_cast<const v8bf*>(p + 8);
  return __builtin_shufflevector(lo, hi, 0,1,2,3,4,5,6,7,8,9,10,11,12,13,14,15);
}

// -------------------- stage 1: GEMM + bias + relu + chunk-local top-64 -----
__global__ __launch_bounds__(256) void gemm_topk_stage1(
    const __bf16* __restrict__ xb, const __bf16* __restrict__ Wb,
    const float* __restrict__ bias,
    float* __restrict__ cand_v, int* __restrict__ cand_i) {
  __shared__ float acts[M_TILE][N_STEP + 1];   // +1 pad: avoid bank conflicts
  __shared__ float topv[M_TILE][KSEL];
  __shared__ int   topi[M_TILE][KSEL];

  const int tid  = threadIdx.x;
  const int lane = tid & 31;
  const int wave = tid >> 5;                 // 8 waves
  const int m0   = blockIdx.x * M_TILE;      // M-tile fastest -> W chunk L2 reuse
  const int chunk = blockIdx.y;
  const int f_chunk = chunk * N_CHUNK;

  const int l15   = lane & 15;
  const int hi    = lane >> 4;
  const int kofsA = hi * 16;                 // A per-lane K offset
  const int kofsB = hi * 16;                 // B per-lane K offset

  const __bf16* arow0 = xb + (size_t)(m0 + l15) * DM + kofsA;       // rows m0..+15
  const __bf16* arow1 = arow0 + (size_t)16 * DM;                    // rows m0+16..+31

  float minval = NEG_INF;
  int   minpos = 0;
  if (tid < M_TILE) {
    for (int q = 0; q < KSEL; ++q) { topv[tid][q] = NEG_INF; topi[tid][q] = 0; }
  }

  for (int ns = 0; ns < N_SUB; ++ns) {
    const int fwave = f_chunk + ns * N_STEP + wave * 32;
    const __bf16* brow0 = Wb + (size_t)(fwave + l15) * DM + kofsB;  // feats f..f+15
    const __bf16* brow1 = brow0 + (size_t)16 * DM;                  // feats f+16..f+31

    v8f c00 = {}, c01 = {}, c10 = {}, c11 = {};
    for (int k0 = 0; k0 < DM; k0 += 32) {
      __builtin_prefetch(brow0 + k0 + 512, 0, 3);   // stream-ahead on W
      __builtin_prefetch(brow1 + k0 + 512, 0, 3);
      v16bf a0 = load_a_frag(arow0 + k0);
      v16bf a1 = load_a_frag(arow1 + k0);
      v16bf b0 = load_b_frag(brow0 + k0);
      v16bf b1 = load_b_frag(brow1 + k0);
      c00 = __builtin_amdgcn_wmma_f32_16x16x32_bf16(false, a0, false, b0, (short)0, c00, false, false);
      c01 = __builtin_amdgcn_wmma_f32_16x16x32_bf16(false, a0, false, b1, (short)0, c01, false, false);
      c10 = __builtin_amdgcn_wmma_f32_16x16x32_bf16(false, a1, false, b0, (short)0, c10, false, false);
      c11 = __builtin_amdgcn_wmma_f32_16x16x32_bf16(false, a1, false, b1, (short)0, c11, false, false);
    }

    // bias + relu, scatter C tiles into LDS acts[32][256]
    const float bias0 = bias[fwave + l15];
    const float bias1 = bias[fwave + 16 + l15];
    const int ncol0 = wave * 32 + l15;
    const int ncol1 = ncol0 + 16;
    const int mlo = hi * 8;                  // C layout: lanes 0-15 M=r, 16-31 M=8+r
    #pragma unroll
    for (int r = 0; r < 8; ++r) {
      const int m = mlo + r;
      acts[m][ncol0]      = fmaxf(c00[r] + bias0, 0.0f);
      acts[m][ncol1]      = fmaxf(c01[r] + bias1, 0.0f);
      acts[m + 16][ncol0] = fmaxf(c10[r] + bias0, 0.0f);
      acts[m + 16][ncol1] = fmaxf(c11[r] + bias1, 0.0f);
    }
    __syncthreads();

    // per-token running top-64 (thread t owns token t; threshold + replace-min)
    if (tid < M_TILE) {
      const int fb = f_chunk + ns * N_STEP;
      for (int j = 0; j < N_STEP; ++j) {
        float v = acts[tid][j];
        if (v > minval) {
          topv[tid][minpos] = v;
          topi[tid][minpos] = fb + j;
          minval = topv[tid][0]; minpos = 0;
          for (int q = 1; q < KSEL; ++q) {
            float tv = topv[tid][q];
            if (tv < minval) { minval = tv; minpos = q; }
          }
        }
      }
    }
    __syncthreads();
  }

  // write chunk-local candidates: cand[token][chunk][KSEL]
  for (int t = tid; t < M_TILE * KSEL; t += 256) {
    const int m = t >> 6, q = t & 63;
    const size_t o = ((size_t)(m0 + m) * CHUNKS + chunk) * KSEL + q;
    cand_v[o] = topv[m][q];
    cand_i[o] = topi[m][q];
  }
}

// -------------------- stage 2: per-token merge of 2048 candidates ----------
__global__ __launch_bounds__(256) void topk_merge_stage2(
    const float* __restrict__ cand_v, const int* __restrict__ cand_i,
    float* __restrict__ out_v, int* __restrict__ out_i) {
  __shared__ float sv[CHUNKS * KSEL];   // 2048
  __shared__ int   si[CHUNKS * KSEL];
  __shared__ float rv[256];
  __shared__ int   rp[256];

  const int tok = blockIdx.x;
  const float* c0 = cand_v + (size_t)tok * CHUNKS * KSEL;
  const int*   i0 = cand_i + (size_t)tok * CHUNKS * KSEL;
  for (int t = threadIdx.x; t < CHUNKS * KSEL; t += 256) { sv[t] = c0[t]; si[t] = i0[t]; }
  __syncthreads();

  for (int sel = 0; sel < KSEL; ++sel) {
    float bv = NEG_INF; int bp = -1; int bi = 0x7fffffff;
    for (int t = threadIdx.x; t < CHUNKS * KSEL; t += 256) {
      float v = sv[t]; int idx = si[t];
      if (v > bv || (v == bv && idx < bi)) { bv = v; bp = t; bi = idx; }
    }
    rv[threadIdx.x] = bv; rp[threadIdx.x] = bp;
    __syncthreads();
    for (int s = 128; s > 0; s >>= 1) {
      if (threadIdx.x < s) {
        float v1 = rv[threadIdx.x];     int p1 = rp[threadIdx.x];
        float v2 = rv[threadIdx.x + s]; int p2 = rp[threadIdx.x + s];
        int i1 = (p1 >= 0) ? si[p1] : 0x7fffffff;
        int i2 = (p2 >= 0) ? si[p2] : 0x7fffffff;
        if (v2 > v1 || (v2 == v1 && i2 < i1)) { rv[threadIdx.x] = v2; rp[threadIdx.x] = p2; }
      }
      __syncthreads();
    }
    if (threadIdx.x == 0) {
      const int p = rp[0];
      out_v[(size_t)tok * KSEL + sel] = rv[0];
      out_i[(size_t)tok * KSEL + sel] = si[p];
      sv[p] = NEG_INF;                       // knock out the winner
    }
    __syncthreads();
  }
}

// ---------------------------------------------------------------------------
extern "C" void kernel_launch(void* const* d_in, const int* in_sizes, int n_in,
                              void* d_out, int out_size, void* d_ws, size_t ws_size,
                              hipStream_t stream) {
  (void)in_sizes; (void)n_in; (void)out_size; (void)ws_size;
  const float* x = (const float*)d_in[0];   // (1, 2048, 2048)
  const float* W = (const float*)d_in[1];   // (65536, 2048)
  const float* b = (const float*)d_in[2];   // (65536,)
  // d_in[3] = k (==64, compile-time constant here)

  // workspace layout: Wb (256MB) | xb (8MB) | cand_v (16MB) | cand_i (16MB)
  char* ws = (char*)d_ws;
  __bf16* Wb = (__bf16*)ws;
  __bf16* xb = (__bf16*)(ws + (size_t)NF * DM * 2);
  float* cand_v = (float*)(ws + (size_t)NF * DM * 2 + (size_t)S * DM * 2);
  int*   cand_i = (int*)((char*)cand_v + (size_t)S * CHUNKS * KSEL * sizeof(float));

  const long nW = (long)NF * DM;   // 134,217,728
  const long nX = (long)S * DM;    //   4,194,304
  cvt_f32_bf16<<<(unsigned)(nW / (8 * 256)), 256, 0, stream>>>(W, Wb, nW);
  cvt_f32_bf16<<<(unsigned)(nX / (8 * 256)), 256, 0, stream>>>(x, xb, nX);

  dim3 grid1(S / M_TILE, CHUNKS);  // (64, 32): M fastest -> W chunk stays in L2
  gemm_topk_stage1<<<grid1, 256, 0, stream>>>(xb, Wb, b, cand_v, cand_i);

  float* out_v = (float*)d_out;
  int*   out_i = (int*)d_out + (size_t)S * KSEL;   // idx stored as raw int32
  topk_merge_stage2<<<S, 256, 0, stream>>>(cand_v, cand_i, out_v, out_i);
}
